// GraphSagePoolAggregator_81527069213082
// MI455X (gfx1250) — matrix operand order repaired and checked by reading, not compiled, not run on hardware
//
#include <hip/hip_runtime.h>

typedef __attribute__((ext_vector_type(16))) _Float16 v16h;
typedef __attribute__((ext_vector_type(8)))  _Float16 v8h;
typedef __attribute__((ext_vector_type(8)))  float    v8f;

#define N_NODES 12288
#define NFEATS  256
#define NHID    128
#define SUPW    144   // NHID + 16 pad rows: row 128 = ones (degree), 129..143 = 0
#define BM      128   // output rows per workgroup (8 waves x 16)
#define BK      32    // K per step (WMMA f16 K)
#define PITCH   40    // f16 LDS pitch: 80 bytes, multiple of 16B for b128 loads
#define KITERS  (N_NODES / BK)   // 384 total k-steps

// ---------------------------------------------------------------------------
// Kernel 1: support = relu(input @ W^T + b), stored f16 TRANSPOSED: supT[h][k]
// Grid: N/16 blocks x 256 threads; wave w handles h-block w (8 x 16 = 128 cols)
// ---------------------------------------------------------------------------
__global__ __launch_bounds__(256)
void support_kernel(const float* __restrict__ inp, const float* __restrict__ W,
                    const float* __restrict__ b, _Float16* __restrict__ supT) {
  const int tid  = threadIdx.x;
  const int lane = tid & 31;
  const int wave = tid >> 5;           // h-block 0..7
  const int m0   = blockIdx.x * 16;
  const int mrow = lane & 15;
  const int hi   = lane >> 4;          // lane half 0/1
  const int kbA  = hi * 8;             // A K sub-base
  const int kbB  = hi * 16;            // B K half
  const int n    = wave * 16 + mrow;   // hidden column 0..127

  v8f acc = {};
  for (int kb = 0; kb < NFEATS; kb += BK) {
    v16h a, bf;
    // A fragment: input[m0+mrow][kb + kbB .. kb + kbB + 15]
    const float* arow = inp + (size_t)(m0 + mrow) * NFEATS + kb;
#pragma unroll
    for (int e = 0; e < 8; ++e) {
      a[e]     = (_Float16)arow[kbB + e];
      a[e + 8] = (_Float16)arow[kbB + 8 + e];
    }
    // B fragment: B[k][n] = W[n][k] -> contiguous 16 floats in W row
    const float* wrow = W + (size_t)n * NFEATS + kb + kbB;
#pragma unroll
    for (int e = 0; e < 16; ++e) bf[e] = (_Float16)wrow[e];

    acc = __builtin_amdgcn_wmma_f32_16x16x32_f16(false, a, false, bf,
                                                 (short)0, acc, false, false);
  }
  // C rows for this lane are m0+8*hi .. m0+8*hi+7 -> contiguous in k-major supT
  const float bias = b[n];
  v8h st;
#pragma unroll
  for (int r = 0; r < 8; ++r) {
    float v = acc[r] + bias;
    st[r] = (_Float16)(v > 0.0f ? v : 0.0f);   // ReLU
  }
  *(v8h*)(supT + (size_t)n * N_NODES + m0 + 8 * hi) = st;  // one b128 store
}

// ---------------------------------------------------------------------------
// Kernel 2: pad rows of supT: row 128 = 1.0 (degree accumulator), 129..143 = 0
// Contiguous region of 16*N f16 starting at supT + 128*N.
// ---------------------------------------------------------------------------
__global__ void pad_kernel(_Float16* __restrict__ supT) {
  const int idx = blockIdx.x * blockDim.x + threadIdx.x;   // 0..24575
  if (idx >= (16 * N_NODES) / 8) return;
  const int off = idx * 8;
  const _Float16 val = (off < N_NODES) ? (_Float16)1.0f : (_Float16)0.0f;
  v8h v = {val, val, val, val, val, val, val, val};
  *(v8h*)(supT + (size_t)NHID * N_NODES + off) = v;
}

// ---------------------------------------------------------------------------
// Core k-loop (shared by split and fused aggregate kernels).
// acc[0..7] = output tiles, acc[8] = degree tile (ones column of supT).
// ---------------------------------------------------------------------------
__device__ __forceinline__
void aggregate_loop(const float* __restrict__ adj, const _Float16* __restrict__ supT,
                    _Float16* ldsA, _Float16* ldsB, v8f acc[9],
                    int i0, int itBase, int iters,
                    int tid, int wave, int mrow, int kbA, int kbB) {
  for (int it = 0; it < iters; ++it) {
    const int k0 = (itBase + it) * BK;
    // Prefetch next 16KB adj block: one 64B chunk per thread covers 32x512B.
    if (it + 1 < iters) {
      const float* pf = adj + (size_t)(k0 + BK + (tid >> 3)) * N_NODES
                            + i0 + ((tid & 7) << 4);
      __builtin_prefetch(pf, 0, 1);
    }
    __syncthreads();
    // Stage adj [k0..k0+31][i0..i0+127]: float4 loads, binarize, transpose.
#pragma unroll
    for (int j = 0; j < 4; ++j) {
      const int idx = tid + 256 * j;        // 0..1023
      const int r   = idx >> 5;             // k within block
      const int c   = (idx & 31) << 2;      // m within block (x4)
      const float4 v = *(const float4*)(adj + (size_t)(k0 + r) * N_NODES + i0 + c);
      ldsA[(c + 0) * PITCH + r] = (v.x > 0.0f) ? (_Float16)1.0f : (_Float16)0.0f;
      ldsA[(c + 1) * PITCH + r] = (v.y > 0.0f) ? (_Float16)1.0f : (_Float16)0.0f;
      ldsA[(c + 2) * PITCH + r] = (v.z > 0.0f) ? (_Float16)1.0f : (_Float16)0.0f;
      ldsA[(c + 3) * PITCH + r] = (v.w > 0.0f) ? (_Float16)1.0f : (_Float16)0.0f;
    }
    // Stage supT tile [0..143][k0..k0+31]: pure 16B global->LDS copies.
#pragma unroll
    for (int j = 0; j < 3; ++j) {
      const int idx = tid + 256 * j;        // 0..767, use 576
      if (idx < 576) {
        const int n = idx >> 2;
        const int q = (idx & 3) << 3;       // f16 offset within row
        *(v8h*)(ldsB + n * PITCH + q) =
            *(const v8h*)(supT + (size_t)n * N_NODES + k0 + q);
      }
    }
    __syncthreads();

    // A fragment: lane holds row m, K = {kbB..kbB+15}
    const _Float16* ap = &ldsA[(wave * 16 + mrow) * PITCH];
    const v8h alo = *(const v8h*)(ap + kbB);
    const v8h ahi = *(const v8h*)(ap + kbB + 8);
    const v16h a = __builtin_shufflevector(alo, ahi,
        0, 1, 2, 3, 4, 5, 6, 7, 8, 9, 10, 11, 12, 13, 14, 15);

#pragma unroll
    for (int nb = 0; nb < 9; ++nb) {
      const _Float16* bp = &ldsB[(nb * 16 + mrow) * PITCH + kbB];
      const v8h blo = *(const v8h*)(bp);
      const v8h bhi = *(const v8h*)(bp + 8);
      const v16h bf = __builtin_shufflevector(blo, bhi,
          0, 1, 2, 3, 4, 5, 6, 7, 8, 9, 10, 11, 12, 13, 14, 15);
      acc[nb] = __builtin_amdgcn_wmma_f32_16x16x32_f16(false, a, false, bf,
                                                       (short)0, acc[nb],
                                                       false, false);
    }
  }
}

// ---------------------------------------------------------------------------
// Kernel 3a: split-K aggregation. Grid (N/BM, KS). Each block writes its
// partial output tile + partial degree to workspace (non-atomic, deterministic).
// ---------------------------------------------------------------------------
__global__ __launch_bounds__(256)
void aggregate_split(const float* __restrict__ adj, const _Float16* __restrict__ supT,
                     float* __restrict__ pOut, float* __restrict__ pDeg,
                     int itersPerSplit) {
  __shared__ __align__(16) _Float16 ldsA[BM * PITCH];
  __shared__ __align__(16) _Float16 ldsB[SUPW * PITCH];

  const int tid  = threadIdx.x;
  const int lane = tid & 31;
  const int wave = tid >> 5;
  const int i0   = blockIdx.x * BM;
  const int s    = blockIdx.y;
  const int mrow = lane & 15;
  const int hi   = lane >> 4;

  v8f acc[9];
#pragma unroll
  for (int t = 0; t < 9; ++t) acc[t] = (v8f){};

  aggregate_loop(adj, supT, ldsA, ldsB, acc, i0, s * itersPerSplit, itersPerSplit,
                 tid, wave, mrow, hi * 8, hi * 16);

  // Partial degree from tile 8 column 0 (lanes with mrow==0)
  if (mrow == 0) {
#pragma unroll
    for (int r = 0; r < 8; ++r)
      pDeg[(size_t)s * N_NODES + i0 + wave * 16 + r + 8 * hi] = acc[8][r];
  }
#pragma unroll
  for (int r = 0; r < 8; ++r) {
    const int m = i0 + wave * 16 + r + 8 * hi;
    float* o = pOut + ((size_t)s * N_NODES + m) * NHID + mrow;
#pragma unroll
    for (int nb = 0; nb < 8; ++nb) o[nb * 16] = acc[nb][r];
  }
}

// ---------------------------------------------------------------------------
// Kernel 3b: fixed-order reduction over splits + degree normalization.
// ---------------------------------------------------------------------------
__global__ void finalize_kernel(const float* __restrict__ pOut,
                                const float* __restrict__ pDeg,
                                float* __restrict__ out, int ks) {
  const int idx = blockIdx.x * blockDim.x + threadIdx.x;
  if (idx >= N_NODES * NHID) return;
  const int m = idx >> 7;
  float sum = 0.0f, deg = 0.0f;
  for (int s = 0; s < ks; ++s) {
    sum += pOut[(size_t)s * N_NODES * NHID + idx];
    deg += pDeg[(size_t)s * N_NODES + m];
  }
  out[idx] = sum / deg;
}

// ---------------------------------------------------------------------------
// Kernel 3c: fused fallback (no workspace partials): single pass, degree
// broadcast via LDS, scaled store directly to out.
// ---------------------------------------------------------------------------
__global__ __launch_bounds__(256)
void aggregate_fused(const float* __restrict__ adj, const _Float16* __restrict__ supT,
                     float* __restrict__ out) {
  __shared__ __align__(16) _Float16 ldsA[BM * PITCH];
  __shared__ __align__(16) _Float16 ldsB[SUPW * PITCH];
  __shared__ float ldsDeg[8 * 16];

  const int tid  = threadIdx.x;
  const int lane = tid & 31;
  const int wave = tid >> 5;
  const int i0   = blockIdx.x * BM;
  const int mrow = lane & 15;
  const int hi   = lane >> 4;

  v8f acc[9];
#pragma unroll
  for (int t = 0; t < 9; ++t) acc[t] = (v8f){};

  aggregate_loop(adj, supT, ldsA, ldsB, acc, i0, 0, KITERS,
                 tid, wave, mrow, hi * 8, hi * 16);

  if (mrow == 0) {
#pragma unroll
    for (int r = 0; r < 8; ++r) ldsDeg[wave * 16 + hi * 8 + r] = acc[8][r];
  }
  __syncthreads();
#pragma unroll
  for (int r = 0; r < 8; ++r) {
    const float scale = 1.0f / ldsDeg[wave * 16 + hi * 8 + r];
    const int m = i0 + wave * 16 + r + 8 * hi;
#pragma unroll
    for (int nb = 0; nb < 8; ++nb)
      out[(size_t)m * NHID + nb * 16 + mrow] = acc[nb][r] * scale;
  }
}

// ---------------------------------------------------------------------------
extern "C" void kernel_launch(void* const* d_in, const int* in_sizes, int n_in,
                              void* d_out, int out_size, void* d_ws, size_t ws_size,
                              hipStream_t stream) {
  const float* inp = (const float*)d_in[0];  // [N, NFEATS]
  const float* adj = (const float*)d_in[1];  // [N, N]
  const float* W   = (const float*)d_in[2];  // [NHID, NFEATS]
  const float* b   = (const float*)d_in[3];  // [NHID]
  float* out       = (float*)d_out;          // [N, NHID]
  _Float16* supT   = (_Float16*)d_ws;        // [SUPW, N] f16 (~3.54 MB)

  (void)in_sizes; (void)n_in; (void)out_size;

  const size_t SUP_B = (size_t)SUPW * N_NODES * sizeof(_Float16);
  const size_t OUT_B = (size_t)N_NODES * NHID * sizeof(float);
  const size_t DEG_B = (size_t)N_NODES * sizeof(float);

  // Pick largest split-K whose partials fit in the workspace (deterministic).
  int ks = 0;
  for (int c = 8; c >= 1; c >>= 1) {
    if (SUP_B + (size_t)c * (OUT_B + DEG_B) <= ws_size) { ks = c; break; }
  }

  support_kernel<<<N_NODES / 16, 256, 0, stream>>>(inp, W, b, supT);
  pad_kernel<<<(16 * N_NODES / 8 + 255) / 256, 256, 0, stream>>>(supT);

  if (ks > 0) {
    float* pOut = (float*)((char*)d_ws + SUP_B);
    float* pDeg = (float*)((char*)d_ws + SUP_B + (size_t)ks * OUT_B);
    aggregate_split<<<dim3(N_NODES / BM, ks), 256, 0, stream>>>(
        adj, supT, pOut, pDeg, KITERS / ks);
    finalize_kernel<<<(N_NODES * NHID + 255) / 256, 256, 0, stream>>>(
        pOut, pDeg, out, ks);
  } else {
    aggregate_fused<<<N_NODES / BM, 256, 0, stream>>>(adj, supT, out);
  }
}
